// DenseCRFLoss_73701638800093
// MI455X (gfx1250) — compile-verified, matrix operand order checked by reading, not run on hardware
//
#include <hip/hip_runtime.h>
#include <hip/hip_bf16.h>

typedef __attribute__((ext_vector_type(16))) _Float16 v16h;
typedef __attribute__((ext_vector_type(8)))  _Float16 v8h;
typedef __attribute__((ext_vector_type(8)))  float    v8f;

#define N_BATCH   4
#define K_SEG     4
#define H_FULL    128
#define W_FULL    128
#define HS        64
#define WS_DIM    64
#define P_PTS     (HS * WS_DIM)      // 4096 points per batch
#define N_TILES   (P_PTS / 16)       // 256 16-row tiles
#define SIGMA_RGB    15.0f
#define SIGMA_XY_EFF 50.0f           // SIGMA_XY * SCALE = 100 * 0.5
#define CRF_WEIGHT   1e-7f

// ---------------------------------------------------------------------------
// Prep: build padded f16 feature rows [N][P][8] = {x,y,r,g,b,0,0,0},
//       padded f16 seg rows           [N][P][8] = {s0..s3,0,0,0,0},
//       f32 squared feature norms     [N][P].
// ---------------------------------------------------------------------------
__global__ void crf_prep_kernel(const float* __restrict__ images,
                                const float* __restrict__ segs,
                                _Float16* __restrict__ F,
                                _Float16* __restrict__ Sg,
                                float*    __restrict__ sq) {
  int idx = blockIdx.x * blockDim.x + threadIdx.x;
  if (idx >= N_BATCH * P_PTS) return;
  int b = idx >> 12;             // / P_PTS
  int p = idx & (P_PTS - 1);
  int i = p >> 6;                // row in downsampled grid
  int j = p & (WS_DIM - 1);      // col

  const int HW = H_FULL * W_FULL;
  int ibase = ((b * 3 + 0) * H_FULL + 2 * i) * W_FULL + 2 * j;
  float r  = images[ibase];
  float g  = images[ibase + HW];
  float bl = images[ibase + 2 * HW];

  float fx = (float)j / SIGMA_XY_EFF;
  float fy = (float)i / SIGMA_XY_EFF;
  float fr = r  / SIGMA_RGB;
  float fg = g  / SIGMA_RGB;
  float fb = bl / SIGMA_RGB;

  v8h f;
  f[0] = (_Float16)fx; f[1] = (_Float16)fy;
  f[2] = (_Float16)fr; f[3] = (_Float16)fg; f[4] = (_Float16)fb;
  f[5] = (_Float16)0.0f; f[6] = (_Float16)0.0f; f[7] = (_Float16)0.0f;
  ((v8h*)F)[idx] = f;

  sq[idx] = fx * fx + fy * fy + fr * fr + fg * fg + fb * fb;

  v8h s;
  int sbase = ((b * K_SEG + 0) * H_FULL + 2 * i) * W_FULL + 2 * j;
#pragma unroll
  for (int k = 0; k < K_SEG; ++k) {
    const float* sp = segs + sbase + k * HW;
    float m = 0.25f * (sp[0] + sp[1] + sp[W_FULL] + sp[W_FULL + 1]);
    s[k] = (_Float16)m;
  }
  s[4] = (_Float16)0.0f; s[5] = (_Float16)0.0f;
  s[6] = (_Float16)0.0f; s[7] = (_Float16)0.0f;
  ((v8h*)Sg)[idx] = s;
}

__global__ void crf_zero_out_kernel(float* out) {
  if (threadIdx.x == 0 && blockIdx.x == 0) out[0] = 0.0f;
}

// Build a 16-bit WMMA A/B fragment (ISA §7.12.2): lanes 0-15 hold K=0..7 in
// the first 8 halves (our only nonzero K slots); lanes 16-31 hold K=8..15 and
// K=24..31 which are all zero-padded -> entire fragment zero on those lanes.
__device__ __forceinline__ v16h crf_load_frag(const v8h* __restrict__ rows,
                                              int row, bool lane_lo) {
  v16h f;
#pragma unroll
  for (int i = 0; i < 16; ++i) f[i] = (_Float16)0.0f;
  if (lane_lo) {
    v8h d = rows[row];
#pragma unroll
    for (int i = 0; i < 8; ++i) f[i] = d[i];
  }
  return f;
}

// ---------------------------------------------------------------------------
// Main: one wave32 per (batch, p-tile). A fragments hoisted; loop over all
// 256 q-tiles doing 2x v_wmma_f32_16x16x32_f16 + 8-wide exp epilogue.
// ---------------------------------------------------------------------------
__global__ void crf_wmma_kernel(const _Float16* __restrict__ F,
                                const _Float16* __restrict__ Sg,
                                const float*    __restrict__ sq,
                                float*          __restrict__ out) {
  int wave = (int)((blockIdx.x * blockDim.x + threadIdx.x) >> 5);
  int lane = (int)(threadIdx.x & 31);
  int b    = wave >> 8;            // / N_TILES
  int pt   = wave & (N_TILES - 1);
  int p0   = pt << 4;

  const v8h*   Fb  = (const v8h*)F  + (size_t)b * P_PTS;
  const v8h*   Sb  = (const v8h*)Sg + (size_t)b * P_PTS;
  const float* sqb = sq + (size_t)b * P_PTS;

  bool lane_lo = lane < 16;
  int  nq      = lane & 15;

  // A fragments: rows p0..p0+15 (reused across the whole q loop).
  v16h aF = crf_load_frag(Fb, p0 + nq, lane_lo);
  v16h aS = crf_load_frag(Sb, p0 + nq, lane_lo);

  // C/D layout: lane l, VGPR v -> M = v + 8*(l>>4), N = l&15.
  float sqP[8];
  int mbase = p0 + ((lane >> 4) << 3);
#pragma unroll
  for (int v = 0; v < 8; ++v) sqP[v] = sqb[mbase + v];

  float acc = 0.0f;
  for (int qt = 0; qt < N_TILES; ++qt) {
    int q0 = qt << 4;
    if (qt + 1 < N_TILES) {  // pull next q-tile rows toward L2/L0
      __builtin_prefetch((const void*)(Fb + q0 + 16 + nq), 0, 1);
      __builtin_prefetch((const void*)(Sb + q0 + 16 + nq), 0, 1);
    }
    v16h bF = crf_load_frag(Fb, q0 + nq, lane_lo);
    v16h bS = crf_load_frag(Sb, q0 + nq, lane_lo);

    v8f cF, cS;
#pragma unroll
    for (int v = 0; v < 8; ++v) { cF[v] = 0.0f; cS[v] = 0.0f; }

    // cross-feature gram: f_p . f_q  (K padded 5 -> 32)
    cF = __builtin_amdgcn_wmma_f32_16x16x32_f16(
        /*neg_a=*/false, aF, /*neg_b=*/false, bF,
        /*c_mod=*/(short)0, cF, /*reuse_a=*/false, /*reuse_b=*/false);
    // seg gram: <S_p, S_q>  (K padded 4 -> 32)
    cS = __builtin_amdgcn_wmma_f32_16x16x32_f16(
        /*neg_a=*/false, aS, /*neg_b=*/false, bS,
        /*c_mod=*/(short)0, cS, /*reuse_a=*/false, /*reuse_b=*/false);

    float sqQ = sqb[q0 + nq];
#pragma unroll
    for (int v = 0; v < 8; ++v) {
      float d2 = fmaxf(sqP[v] + sqQ - 2.0f * cF[v], 0.0f);
      acc += __expf(-0.5f * d2) * cS[v];
    }
  }

  // wave32 reduction, then one atomic per wave.
#pragma unroll
  for (int off = 16; off > 0; off >>= 1) acc += __shfl_down(acc, off, 32);
  if (lane == 0) {
    atomicAdd(out, acc * (-CRF_WEIGHT / (float)N_BATCH));
  }
}

extern "C" void kernel_launch(void* const* d_in, const int* in_sizes, int n_in,
                              void* d_out, int out_size, void* d_ws, size_t ws_size,
                              hipStream_t stream) {
  const float* images = (const float*)d_in[0];          // (4,3,128,128) f32
  const float* segs   = (const float*)d_in[1];          // (4,4,128,128) f32
  float* out = (float*)d_out;                           // scalar f32

  char* ws = (char*)d_ws;
  const size_t fbytes = (size_t)N_BATCH * P_PTS * 8 * sizeof(_Float16); // 256 KB
  _Float16* F  = (_Float16*)(ws);
  _Float16* Sg = (_Float16*)(ws + fbytes);
  float*    sq = (float*)(ws + 2 * fbytes);             // +64 KB

  crf_zero_out_kernel<<<1, 64, 0, stream>>>(out);

  int npts = N_BATCH * P_PTS;
  crf_prep_kernel<<<(npts + 255) / 256, 256, 0, stream>>>(images, segs, F, Sg, sq);

  int nthreads = N_BATCH * N_TILES * 32;                // 1 wave per p-tile
  crf_wmma_kernel<<<(nthreads + 255) / 256, 256, 0, stream>>>(F, Sg, sq, out);
}
